// FuseEfficientAttention_21096879358322
// MI455X (gfx1250) — matrix operand order repaired and checked by reading, not compiled, not run on hardware
//
#include <hip/hip_runtime.h>
#include <hip/hip_bf16.h>

// ---------------------------------------------------------------------------
// Types for CDNA5 WMMA (wave32): V_WMMA_F32_16X16X32_BF16
// ---------------------------------------------------------------------------
typedef __attribute__((ext_vector_type(16))) __bf16 v16bf;
typedef __attribute__((ext_vector_type(8)))  float  v8f;

#define THREADS 256

// ---------------------------------------------------------------------------
// fp32 <-> split-bf16 helpers. A "pair" packs hi bf16 in bits[15:0] and the
// residual lo bf16 in bits[31:16] (same 4B footprint as the fp32 it replaces).
// ---------------------------------------------------------------------------
union BfBits { unsigned short s; __bf16 b; };

static __device__ __forceinline__ __bf16 bits2bf(unsigned short s) {
    BfBits u; u.s = s; return u.b;
}
static __device__ __forceinline__ float bfbits2f(unsigned short s) {
    return __uint_as_float(((unsigned)s) << 16);
}
static __device__ __forceinline__ unsigned short f2bfbits_rne(float f) {
    unsigned u = __float_as_uint(f);
    unsigned r = u + 0x7FFFu + ((u >> 16) & 1u);
    return (unsigned short)(r >> 16);
}
// split fp32 into hi+lo bf16 (hi+lo ~= f to ~16 mantissa bits), packed
static __device__ __forceinline__ unsigned pack_split(float f) {
    unsigned short hs = f2bfbits_rne(f);
    float rem = f - bfbits2f(hs);
    unsigned short ls = f2bfbits_rne(rem);
    return (unsigned)hs | ((unsigned)ls << 16);
}
static __device__ __forceinline__ void unpack_pair(unsigned u, __bf16& hi, __bf16& lo) {
    hi = bits2bf((unsigned short)u);
    lo = bits2bf((unsigned short)(u >> 16));
}

// ---------------------------------------------------------------------------
// WMMA fragment loaders (ISA 7.12.2 layouts, wave32)
// A 16x32 bf16: lanes 0-15 row M=lane, K {kh*8+0..7} in V0-3, {16+kh*8+0..7} in V4-7
// ---------------------------------------------------------------------------
static __device__ __forceinline__ v16bf load_frag_a(const __bf16* lds, int lda) {
    const int lane = threadIdx.x & 31;
    const int row  = lane & 15;
    const int kh   = lane >> 4;
    const __bf16* base = lds + row * lda;
    v16bf a;
#pragma unroll
    for (int e = 0; e < 8; ++e) a[e] = base[kh * 8 + e];
#pragma unroll
    for (int e = 0; e < 8; ++e) a[e + 8] = base[16 + kh * 8 + e];
    return a;
}
// B 32x16 bf16 (LDS stored [N][K]): lanes 0-15 col N=lane, K {kh*8+0..7} in V0-3,
// {16+kh*8+0..7} in V4-7 (same K order as A)
static __device__ __forceinline__ v16bf load_frag_b(const __bf16* lds, int ldb) {
    const int lane = threadIdx.x & 31;
    const int col  = lane & 15;
    const int kh   = lane >> 4;
    const __bf16* base = lds + col * ldb;
    v16bf b;
#pragma unroll
    for (int e = 0; e < 8; ++e) b[e] = base[kh * 8 + e];
#pragma unroll
    for (int e = 0; e < 8; ++e) b[e + 8] = base[16 + kh * 8 + e];
    return b;
}

static __device__ __forceinline__ v8f wmma_bf16(v16bf a, v16bf b, v8f c) {
    return __builtin_amdgcn_wmma_f32_16x16x32_bf16(
        /*neg_a=*/false, a, /*neg_b=*/false, b,
        /*c_mod=*/(short)0, c, /*reuse_a=*/false, /*reuse_b=*/false);
}
// split-precision accumulate: hi*hi + lo*hi + hi*lo (lo*lo negligible)
static __device__ __forceinline__ v8f mma_split(v16bf ah, v16bf al,
                                                v16bf bh, v16bf bl, v8f c) {
    c = wmma_bf16(ah, bh, c);
    c = wmma_bf16(al, bh, c);
    c = wmma_bf16(ah, bl, c);
    return c;
}

// ===========================================================================
// Kernel 0a: split x once into packed pairs (reused by 8 n-tiles per WG-row)
// ===========================================================================
__global__ __launch_bounds__(THREADS) void prep_split_kernel(
    const float* __restrict__ src, unsigned* __restrict__ dst)
{
    const size_t i = (size_t)blockIdx.x * THREADS + threadIdx.x;
    dst[i] = pack_split(src[i]);
}

// ===========================================================================
// Kernel 0b: split + transpose the 4 weight matrices into [n][k] pair form
// blockIdx.y selects Wk/Wq/Wv/Wr; each is 512x512
// ===========================================================================
__global__ __launch_bounds__(THREADS) void prep_wt_kernel(
    const float* __restrict__ W0, const float* __restrict__ W1,
    const float* __restrict__ W2, const float* __restrict__ W3,
    unsigned* __restrict__ T0, unsigned* __restrict__ T1,
    unsigned* __restrict__ T2, unsigned* __restrict__ T3)
{
    const float* W; unsigned* T;
    if (blockIdx.y == 0)      { W = W0; T = T0; }
    else if (blockIdx.y == 1) { W = W1; T = T1; }
    else if (blockIdx.y == 2) { W = W2; T = T2; }
    else                      { W = W3; T = T3; }
    const int i  = blockIdx.x * THREADS + threadIdx.x; // < 512*512
    const int n  = i & 511;
    const int kk = i >> 9;
    T[(size_t)n * 512 + kk] = pack_split(W[(size_t)kk * 512 + n]);
}

// ===========================================================================
// Kernel 1: QKV projections  out = x @ W + b   (M=32768, K=512, N=512)
// A from packed x pairs, B from packed transposed W pairs; K/Q stored fp32
// (softmax wants fp32), V stored directly as packed pairs.
// 256 threads = 8 waves (4M x 2N); WG tile 128x64; wave tile 32x32 (2x2 wmma)
// ===========================================================================
__global__ __launch_bounds__(THREADS) void proj_gemm_kernel(
    const unsigned* __restrict__ xp,
    const unsigned* __restrict__ WkT, const float* __restrict__ bk, float* __restrict__ outK,
    const unsigned* __restrict__ WqT, const float* __restrict__ bq, float* __restrict__ outQ,
    const unsigned* __restrict__ WvT, const float* __restrict__ bv, unsigned* __restrict__ outVp)
{
    const int K = 512, N = 512;
    const unsigned* Wt; const float* bias;
    float* Of = nullptr; unsigned* Op = nullptr;
    bool packout = false;
    if (blockIdx.y == 0)      { Wt = WkT; bias = bk; Of = outK; }
    else if (blockIdx.y == 1) { Wt = WqT; bias = bq; Of = outQ; }
    else                      { Wt = WvT; bias = bv; Op = outVp; packout = true; }

    const int tiles_n = N / 64;                   // 8
    const int m0 = (blockIdx.x / tiles_n) * 128;
    const int n0 = (blockIdx.x % tiles_n) * 64;

    __shared__ __bf16 Ahi[128][40], Alo[128][40];
    __shared__ __bf16 Bhi[64][40],  Blo[64][40];

    const int tid = threadIdx.x;
    const int wid = tid >> 5;
    const int mw  = wid >> 1;  // 0..3
    const int nw  = wid & 1;   // 0..1

    v8f acc[2][2] = {};

    for (int k0 = 0; k0 < K; k0 += 32) {
        if (k0 + 32 < K)
            __builtin_prefetch(&xp[(size_t)(m0 + (tid >> 1)) * K + k0 + 32 + (tid & 1) * 16], 0, 1);
        for (int i = tid; i < 128 * 32; i += THREADS) {
            int r = i >> 5, c = i & 31;
            unpack_pair(xp[(size_t)(m0 + r) * K + (k0 + c)], Ahi[r][c], Alo[r][c]);
        }
        for (int i = tid; i < 64 * 32; i += THREADS) {
            int n = i >> 5, c = i & 31;  // coalesced along k ([n][k] layout)
            unpack_pair(Wt[(size_t)(n0 + n) * K + (k0 + c)], Bhi[n][c], Blo[n][c]);
        }
        __syncthreads();

        v16bf ah[2], al[2], bh[2], bl[2];
#pragma unroll
        for (int im = 0; im < 2; ++im) {
            ah[im] = load_frag_a(&Ahi[mw * 32 + im * 16][0], 40);
            al[im] = load_frag_a(&Alo[mw * 32 + im * 16][0], 40);
        }
#pragma unroll
        for (int in = 0; in < 2; ++in) {
            bh[in] = load_frag_b(&Bhi[nw * 32 + in * 16][0], 40);
            bl[in] = load_frag_b(&Blo[nw * 32 + in * 16][0], 40);
        }
#pragma unroll
        for (int im = 0; im < 2; ++im)
#pragma unroll
            for (int in = 0; in < 2; ++in)
                acc[im][in] = mma_split(ah[im], al[im], bh[in], bl[in], acc[im][in]);
        __syncthreads();
    }

    const int lane = tid & 31;
    const int colL = lane & 15;
    const int rb   = (lane >> 4) * 8;
#pragma unroll
    for (int im = 0; im < 2; ++im)
#pragma unroll
        for (int in = 0; in < 2; ++in) {
            int mbase = m0 + mw * 32 + im * 16 + rb;
            int nbase = n0 + nw * 32 + in * 16 + colL;
            float bb = bias[nbase];
            if (packout) {
#pragma unroll
                for (int r = 0; r < 8; ++r)
                    Op[(size_t)(mbase + r) * N + nbase] = pack_split(acc[im][in][r] + bb);
            } else {
#pragma unroll
                for (int r = 0; r < 8; ++r)
                    Of[(size_t)(mbase + r) * N + nbase] = acc[im][in][r] + bb;
            }
        }
}

// ===========================================================================
// Kernel 2: softmax over contiguous 4096-long rows (keys); reads fp32,
// writes packed split pairs IN PLACE (same 4B slot per element)
// ===========================================================================
__global__ __launch_bounds__(THREADS) void softmax_rows_kernel(float* __restrict__ buf)
{
    const int rowlen = 4096;
    float*    rowf = buf + (size_t)blockIdx.x * rowlen;
    unsigned* rowp = (unsigned*)rowf;
    __shared__ float red[THREADS];
    const int tid = threadIdx.x;

    float m = -3.402823466e38f;
    for (int i = tid; i < rowlen; i += THREADS) m = fmaxf(m, rowf[i]);
    red[tid] = m; __syncthreads();
    for (int s = THREADS / 2; s > 0; s >>= 1) {
        if (tid < s) red[tid] = fmaxf(red[tid], red[tid + s]);
        __syncthreads();
    }
    m = red[0]; __syncthreads();

    float sum = 0.f;
    for (int i = tid; i < rowlen; i += THREADS) sum += expf(rowf[i] - m);
    red[tid] = sum; __syncthreads();
    for (int s = THREADS / 2; s > 0; s >>= 1) {
        if (tid < s) red[tid] += red[tid + s];
        __syncthreads();
    }
    const float inv = 1.0f / red[0];
    for (int i = tid; i < rowlen; i += THREADS) {
        float e = expf(rowf[i] - m) * inv;   // read-before-write, index owned
        rowp[i] = pack_split(e);
    }
}

// ===========================================================================
// Kernel 3: softmax over per-head channel (queries, 64 elems, stride 4096)
// one thread per (b,h,n) column; fp32 in, packed split pairs out, in place
// ===========================================================================
__global__ __launch_bounds__(THREADS) void softmax_cols_kernel(float* __restrict__ buf)
{
    const int idx = blockIdx.x * THREADS + threadIdx.x; // < 8*8*4096
    const int n   = idx & 4095;
    const int bh  = idx >> 12;
    float*    basef = buf + (size_t)bh * 64 * 4096 + n;
    unsigned* basep = (unsigned*)basef;

    float v[64];
    float m = -3.402823466e38f;
#pragma unroll
    for (int k = 0; k < 64; ++k) { v[k] = basef[(size_t)k * 4096]; m = fmaxf(m, v[k]); }
    float s = 0.f;
#pragma unroll
    for (int k = 0; k < 64; ++k) { v[k] = expf(v[k] - m); s += v[k]; }
    const float inv = 1.0f / s;
#pragma unroll
    for (int k = 0; k < 64; ++k) basep[(size_t)k * 4096] = pack_split(v[k] * inv);
}

// ===========================================================================
// Kernel 4: context[bh][k][v] = sum_n ksm[bh][k][n] * vals[bh][v][n]
// operands are packed pairs; one WG per (b,h);
// 8 waves = (2M x 2N spatial) x 2-way K-split, LDS reduce.
// Output written TRANSPOSED as pairs: ctxT[bh][v][k] (ready as A for kernel 5)
// ===========================================================================
__global__ __launch_bounds__(THREADS) void context_gemm_kernel(
    const unsigned* __restrict__ ksm, const unsigned* __restrict__ vals,
    unsigned* __restrict__ ctxT)
{
    const int bh = blockIdx.x;
    const unsigned* Ap = ksm  + (size_t)bh * 64 * 4096;  // [k][n]
    const unsigned* Bp = vals + (size_t)bh * 64 * 4096;  // [v][n]

    __shared__ __bf16 Ahi[64][72], Alo[64][72];
    __shared__ __bf16 Bhi[64][72], Blo[64][72];
    __shared__ float  red[4][32][33];

    const int tid = threadIdx.x;
    const int wid = tid >> 5;
    const int ks  = wid >> 2;        // K-split 0/1
    const int sp  = wid & 3;         // spatial wave
    const int mw  = sp >> 1, nw = sp & 1;

    v8f acc[2][2] = {};

    for (int nb = 0; nb < 4096; nb += 64) {
        for (int i = tid; i < 64 * 64; i += THREADS) {
            int r = i >> 6, c = i & 63;
            unpack_pair(Ap[(size_t)r * 4096 + nb + c], Ahi[r][c], Alo[r][c]);
            unpack_pair(Bp[(size_t)r * 4096 + nb + c], Bhi[r][c], Blo[r][c]);
        }
        __syncthreads();

        const int kc = ks * 32;
        v16bf ah[2], al[2], bh[2], bl[2];
#pragma unroll
        for (int im = 0; im < 2; ++im) {
            ah[im] = load_frag_a(&Ahi[mw * 32 + im * 16][kc], 72);
            al[im] = load_frag_a(&Alo[mw * 32 + im * 16][kc], 72);
        }
#pragma unroll
        for (int in = 0; in < 2; ++in) {
            bh[in] = load_frag_b(&Bhi[nw * 32 + in * 16][kc], 72);
            bl[in] = load_frag_b(&Blo[nw * 32 + in * 16][kc], 72);
        }
#pragma unroll
        for (int im = 0; im < 2; ++im)
#pragma unroll
            for (int in = 0; in < 2; ++in)
                acc[im][in] = mma_split(ah[im], al[im], bh[in], bl[in], acc[im][in]);
        __syncthreads();
    }

    const int lane = tid & 31;
    const int colL = lane & 15;
    const int rb   = (lane >> 4) * 8;
    if (ks == 1) {
#pragma unroll
        for (int im = 0; im < 2; ++im)
#pragma unroll
            for (int in = 0; in < 2; ++in)
#pragma unroll
                for (int r = 0; r < 8; ++r)
                    red[sp][im * 16 + rb + r][in * 16 + colL] = acc[im][in][r];
    }
    __syncthreads();
    if (ks == 0) {
#pragma unroll
        for (int im = 0; im < 2; ++im)
#pragma unroll
            for (int in = 0; in < 2; ++in)
#pragma unroll
                for (int r = 0; r < 8; ++r) {
                    float vsum = acc[im][in][r] + red[sp][im * 16 + rb + r][in * 16 + colL];
                    int k = mw * 32 + im * 16 + rb + r;   // row of context
                    int v = nw * 32 + in * 16 + colL;     // col of context
                    ctxT[(size_t)bh * 4096 + (size_t)v * 64 + k] = pack_split(vsum);
                }
    }
}

// ===========================================================================
// Kernel 5: attended[bh][v][n] = sum_k ctx[bh][k][v] * qsm[bh][k][n]
// A = ctxT pairs [v][k] (already transposed), B = qsm pairs [k][n];
// output written as packed pairs (re-read x8 by kernel 6).
// grid (32 n-tiles, 64 bh); WG 256 = 8 waves (2M x 4N); M=64, K=64, Ntile=128
// ===========================================================================
__global__ __launch_bounds__(THREADS) void attended_gemm_kernel(
    const unsigned* __restrict__ ctxT, const unsigned* __restrict__ qsm,
    unsigned* __restrict__ attp)
{
    const int bh = blockIdx.y;
    const int n0 = blockIdx.x * 128;
    const unsigned* Cp = ctxT + (size_t)bh * 64 * 64;    // [v][k]
    const unsigned* Qp = qsm  + (size_t)bh * 64 * 4096;  // [k][n]
    unsigned*       Og = attp + (size_t)bh * 64 * 4096;  // [v][n]

    __shared__ __bf16 Ahi[64][72],  Alo[64][72];     // A[v][k]
    __shared__ __bf16 Bhi[128][72], Blo[128][72];    // B[n][k]

    const int tid = threadIdx.x;
    const int wid = tid >> 5;
    const int mw  = wid >> 2;  // 0..1
    const int nw  = wid & 3;   // 0..3

    for (int i = tid; i < 64 * 64; i += THREADS) {
        int k = i & 63, v = i >> 6;   // coalesced along k
        unpack_pair(Cp[(size_t)v * 64 + k], Ahi[v][k], Alo[v][k]);
    }
    for (int i = tid; i < 128 * 64; i += THREADS) {
        int n = i & 127, k = i >> 7;  // coalesced along n
        unpack_pair(Qp[(size_t)k * 4096 + n0 + n], Bhi[n][k], Blo[n][k]);
    }
    __syncthreads();

    v8f acc[2][2] = {};
#pragma unroll
    for (int kk = 0; kk < 64; kk += 32) {
        v16bf ah[2], al[2], bh[2], bl[2];
#pragma unroll
        for (int im = 0; im < 2; ++im) {
            ah[im] = load_frag_a(&Ahi[mw * 32 + im * 16][kk], 72);
            al[im] = load_frag_a(&Alo[mw * 32 + im * 16][kk], 72);
        }
#pragma unroll
        for (int in = 0; in < 2; ++in) {
            bh[in] = load_frag_b(&Bhi[nw * 32 + in * 16][kk], 72);
            bl[in] = load_frag_b(&Blo[nw * 32 + in * 16][kk], 72);
        }
#pragma unroll
        for (int im = 0; im < 2; ++im)
#pragma unroll
            for (int in = 0; in < 2; ++in)
                acc[im][in] = mma_split(ah[im], al[im], bh[in], bl[in], acc[im][in]);
    }

    const int lane = tid & 31;
    const int colL = lane & 15;
    const int rb   = (lane >> 4) * 8;
#pragma unroll
    for (int im = 0; im < 2; ++im)
#pragma unroll
        for (int in = 0; in < 2; ++in)
#pragma unroll
            for (int r = 0; r < 8; ++r)
                Og[(size_t)(mw * 32 + im * 16 + rb + r) * 4096
                   + (n0 + nw * 32 + in * 16 + colL)] = pack_split(acc[im][in][r]);
}

// ===========================================================================
// Kernel 6: out = aggregated @ Wr + br,  aggregated[b,n,v] = att[b][v*4096+n]
// A = att pairs (transpose folded into staging), B = WrT pairs
// ===========================================================================
__global__ __launch_bounds__(THREADS) void out_gemm_kernel(
    const unsigned* __restrict__ attp, const unsigned* __restrict__ WrT,
    const float* __restrict__ br, float* __restrict__ out)
{
    const int K = 512, N = 512;
    const int tiles_n = N / 64;                   // 8
    const int m0 = (blockIdx.x / tiles_n) * 128;
    const int n0 = (blockIdx.x % tiles_n) * 64;
    const int b   = m0 >> 12;                     // / 4096 (tiles never cross batch)
    const int nn0 = m0 & 4095;
    const unsigned* Ap = attp + (size_t)b * 512 * 4096;

    __shared__ __bf16 Ahi[128][40], Alo[128][40];
    __shared__ __bf16 Bhi[64][40],  Blo[64][40];

    const int tid = threadIdx.x;
    const int wid = tid >> 5;
    const int mw  = wid >> 1;
    const int nw  = wid & 1;

    v8f acc[2][2] = {};

    for (int k0 = 0; k0 < K; k0 += 32) {
        for (int i = tid; i < 128 * 32; i += THREADS) {
            int r = i & 127, c = i >> 7;  // coalesced along r (= n)
            unpack_pair(Ap[(size_t)(k0 + c) * 4096 + nn0 + r], Ahi[r][c], Alo[r][c]);
        }
        for (int i = tid; i < 64 * 32; i += THREADS) {
            int n = i >> 5, c = i & 31;  // coalesced along k
            unpack_pair(WrT[(size_t)(n0 + n) * K + (k0 + c)], Bhi[n][c], Blo[n][c]);
        }
        __syncthreads();

        v16bf ah[2], al[2], bh[2], bl[2];
#pragma unroll
        for (int im = 0; im < 2; ++im) {
            ah[im] = load_frag_a(&Ahi[mw * 32 + im * 16][0], 40);
            al[im] = load_frag_a(&Alo[mw * 32 + im * 16][0], 40);
        }
#pragma unroll
        for (int in = 0; in < 2; ++in) {
            bh[in] = load_frag_b(&Bhi[nw * 32 + in * 16][0], 40);
            bl[in] = load_frag_b(&Blo[nw * 32 + in * 16][0], 40);
        }
#pragma unroll
        for (int im = 0; im < 2; ++im)
#pragma unroll
            for (int in = 0; in < 2; ++in)
                acc[im][in] = mma_split(ah[im], al[im], bh[in], bl[in], acc[im][in]);
        __syncthreads();
    }

    const int lane = tid & 31;
    const int colL = lane & 15;
    const int rb   = (lane >> 4) * 8;
#pragma unroll
    for (int im = 0; im < 2; ++im)
#pragma unroll
        for (int in = 0; in < 2; ++in) {
            int mbase = m0 + mw * 32 + im * 16 + rb;
            int nbase = n0 + nw * 32 + in * 16 + colL;
            float bb = br[nbase];
#pragma unroll
            for (int r = 0; r < 8; ++r)
                out[(size_t)(mbase + r) * N + nbase] = acc[im][in][r] + bb;
        }
}

// ===========================================================================
// Launch. Inputs: x, Wk, bk, Wq, bq, Wv, bv, Wr, br (all fp32).
// Workspace: xpair | bufK | bufQ | vpair (64MB each) | ctxT (1MB) | 4x WT (1MB)
// bufK lifecycle: K fp32 -> key_sm pairs -> (after context) attended pairs
// ===========================================================================
extern "C" void kernel_launch(void* const* d_in, const int* in_sizes, int n_in,
                              void* d_out, int out_size, void* d_ws, size_t ws_size,
                              hipStream_t stream)
{
    (void)in_sizes; (void)n_in; (void)out_size; (void)ws_size;
    const float* x  = (const float*)d_in[0];
    const float* Wk = (const float*)d_in[1];
    const float* bk = (const float*)d_in[2];
    const float* Wq = (const float*)d_in[3];
    const float* bq = (const float*)d_in[4];
    const float* Wv = (const float*)d_in[5];
    const float* bv = (const float*)d_in[6];
    const float* Wr = (const float*)d_in[7];
    const float* br = (const float*)d_in[8];
    float* out = (float*)d_out;

    const size_t SZ = (size_t)8 * 4096 * 512;  // 16,777,216 elements
    unsigned* xpair = (unsigned*)d_ws;
    float*    bufK  = (float*)(xpair + SZ);
    float*    bufQ  = bufK + SZ;
    unsigned* vpair = (unsigned*)(bufQ + SZ);
    unsigned* ctxT  = vpair + SZ;              // 64*64*64 = 262,144
    unsigned* WkT   = ctxT + (size_t)64 * 64 * 64;
    unsigned* WqT   = WkT + 512 * 512;
    unsigned* WvT   = WqT + 512 * 512;
    unsigned* WrT   = WvT + 512 * 512;

    // 0) one-time splits: x -> pairs; weights -> transposed [n][k] pairs
    prep_split_kernel<<<dim3(65536), THREADS, 0, stream>>>(x, xpair);
    prep_wt_kernel<<<dim3(1024, 4), THREADS, 0, stream>>>(
        Wk, Wq, Wv, Wr, WkT, WqT, WvT, WrT);
    // 1) Q/K/V projections: K,Q fp32; V straight to pairs
    proj_gemm_kernel<<<dim3(2048, 3), THREADS, 0, stream>>>(
        xpair, WkT, bk, bufK, WqT, bq, bufQ, WvT, bv, vpair);
    // 2) key softmax over sequence axis -> pairs in place
    softmax_rows_kernel<<<dim3(4096), THREADS, 0, stream>>>(bufK);
    // 3) query softmax over per-head channel -> pairs in place
    softmax_cols_kernel<<<dim3(1024), THREADS, 0, stream>>>(bufQ);
    // 4) context = key_sm @ values^T per (b,h) -> transposed pairs
    context_gemm_kernel<<<dim3(64), THREADS, 0, stream>>>(
        (const unsigned*)bufK, vpair, ctxT);
    // 5) attended = context^T @ query_sm per (b,h) -> pairs (into bufK)
    attended_gemm_kernel<<<dim3(32, 64), THREADS, 0, stream>>>(
        ctxT, (const unsigned*)bufQ, (unsigned*)bufK);
    // 6) out = aggregated @ Wr + br
    out_gemm_kernel<<<dim3(2048), THREADS, 0, stream>>>(
        (const unsigned*)bufK, WrT, br, out);
}
